// HypNO_39994735460761
// MI455X (gfx1250) — matrix operand order repaired and checked by reading, not compiled, not run on hardware
//
#include <hip/hip_runtime.h>
#include <math.h>

// ---------------------------------------------------------------------------
// HypNO forward for MI455X (gfx1250, wave32, WMMA).
// All 128-wide GEMMs run on v_wmma_f32_16x16x32_f16 with f32 accumulation.
// Neighbor-shifted inputs are handled by A-row index mapping; the scalar
// features (rel_x, du0, |du0|, rel_t, tau) are folded into the epilogue as
// rank-<=3 updates, so no concatenated feature buffers are materialized.
// Weights are pre-swizzled into B-fragment lane order once per launch.
// ---------------------------------------------------------------------------

typedef __attribute__((ext_vector_type(16))) _Float16 v16h;
typedef __attribute__((ext_vector_type(8)))  float    v8f;

#define NXD 256
#define NTD 64
#define BD  4

__device__ __forceinline__ float gelu_f(float v) {
  return 0.5f * v * (1.0f + erff(v * 0.70710678118654752440f));
}

// ------------------------- weight pre-swizzle ------------------------------
// src: row-major [Ktot, 128] f32.  dst: per 128-K block, per 32-K step s,
// per 16-N tile tn: 32 lanes x 16 halves, fragment-ready for v_wmma B operand.
__global__ __launch_bounds__(256) void k_pack(const float* __restrict__ src,
                                              _Float16* __restrict__ dst,
                                              int Ktot) {
  int tid = blockIdx.x * 256 + threadIdx.x;
  if (tid >= Ktot * 128) return;
  int k = tid >> 7, n = tid & 127;
  int blk = k >> 7;          // 128-row K block
  int kk  = k & 127;
  int s   = kk >> 5;         // 32-K step within block
  int r   = kk & 31;
  int lane = (n & 15) | ((r & 8) ? 16 : 0);
  int e = (r & 7) | ((r & 16) >> 1);
  size_t didx = ((size_t)((blk * 4 + s) * 8 + (n >> 4))) * 512 + (size_t)lane * 16 + e;
  dst[didx] = (_Float16)src[(size_t)k * 128 + n];
}

// ------------------------------ utilities ----------------------------------
__global__ __launch_bounds__(256) void k_zero(float* __restrict__ p, long n) {
  long i = (long)blockIdx.x * 256 + threadIdx.x;
  if (i < n) p[i] = 0.f;
}

__global__ __launch_bounds__(256) void k_gelu_ew(const float* __restrict__ in,
                                                 float* __restrict__ out, long n) {
  long i = (long)blockIdx.x * 256 + threadIdx.x;
  if (i < n) out[i] = gelu_f(in[i]);
}

// lift_node layer1:  gelu([u0,x] @ W(2x128) + b)  for M = B*NX rows
__global__ __launch_bounds__(256) void k_lift_node1(const float* __restrict__ u0,
                                                    const float* __restrict__ xg,
                                                    const float* __restrict__ W,
                                                    const float* __restrict__ b,
                                                    float* __restrict__ out) {
  int tid = blockIdx.x * 256 + threadIdx.x;   // 1024*128
  int row = tid >> 7, n = tid & 127;
  int i = row & 255;
  float v = u0[row] * W[n] + xg[i] * W[128 + n] + b[n];
  out[tid] = gelu_f(v);
}

// lift edge/gate layer1: gelu([u0,u_j,dx,|du|] @ W(4x128) + b)
__global__ __launch_bounds__(256) void k_lift_ein1(const float* __restrict__ u0,
                                                   const float* __restrict__ xg,
                                                   const float* __restrict__ W,
                                                   const float* __restrict__ b,
                                                   float* __restrict__ out, int j) {
  int tid = blockIdx.x * 256 + threadIdx.x;   // 1024*128
  int row = tid >> 7, n = tid & 127;
  int i = row & 255, bb = row >> 8;
  int ij = i + j; ij = ij < 0 ? 0 : (ij > 255 ? 255 : ij);
  float e0 = u0[row];
  float e1 = u0[bb * 256 + ij];
  float e2 = xg[ij] - xg[i];
  float e3 = fabsf(e0 - e1);
  float v = e0 * W[n] + e1 * W[128 + n] + e2 * W[256 + n] + e3 * W[384 + n] + b[n];
  out[tid] = gelu_f(v);
}

// tau[64,32] = [sin(t*f), cos(t*f)] @ Wproj(32x32) + bproj   (no activation)
__global__ __launch_bounds__(256) void k_time_tau(const float* __restrict__ tg,
                                                  const float* __restrict__ fr,
                                                  const float* __restrict__ W,
                                                  const float* __restrict__ b,
                                                  float* __restrict__ tau) {
  int tid = blockIdx.x * 256 + threadIdx.x;   // 2048
  if (tid >= NTD * 32) return;
  int tt = tid >> 5, k = tid & 31;
  float tv = tg[tt];
  float s = b[k];
  for (int m = 0; m < 16; ++m) s += sinf(tv * fr[m]) * W[m * 32 + k];
  for (int m = 0; m < 16; ++m) s += cosf(tv * fr[m]) * W[(16 + m) * 32 + k];
  tau[tid] = s;
}

// ttab[64,128] = tau @ time_mix.W0[128:160, :]
__global__ __launch_bounds__(256) void k_ttab(const float* __restrict__ tau,
                                              const float* __restrict__ Wtm,
                                              float* __restrict__ ttab) {
  int tid = blockIdx.x * 256 + threadIdx.x;   // 8192
  if (tid >= NTD * 128) return;
  int tt = tid >> 7, n = tid & 127;
  float s = 0.f;
  for (int m = 0; m < 32; ++m) s += tau[tt * 32 + m] * Wtm[(128 + m) * 128 + n];
  ttab[tid] = s;
}

// wave-per-row 128-dot: mode 0 -> sigmoid gate, mode 1 -> u0 + dot (decoder)
__global__ __launch_bounds__(256) void k_rowdot(const float* __restrict__ Z,
                                                const float* __restrict__ w,
                                                const float* __restrict__ bias,
                                                const float* __restrict__ u0,
                                                float* __restrict__ out,
                                                long M, int mode, int nt) {
  long row = ((long)blockIdx.x * 256 + threadIdx.x) >> 5;
  int lane = threadIdx.x & 31;
  if (row >= M) return;
  const float* zr = Z + row * 128;
  float s = 0.f;
#pragma unroll
  for (int q = 0; q < 4; ++q) { int c = lane + q * 32; s += zr[c] * w[c]; }
#pragma unroll
  for (int off = 16; off > 0; off >>= 1) s += __shfl_xor(s, off, 32);
  if (lane == 0) {
    float v = s + bias[0];
    if (mode == 0) {
      out[row] = 1.f / (1.f + expf(-v));
    } else {
      int i = (int)(row & 255);
      int b = (int)(row / (256l * nt));
      out[row] = u0[b * 256 + i] + v;
    }
  }
}

// ------------------------------ WMMA GEMM ----------------------------------
// Y[M,128] (+)= epilogue( X1@W0 (+ X2@W1) + bias + extras ), K per source = 128.
struct GemmArgs {
  const float* X1;
  const float* X2;         // nullptr -> single source
  const _Float16* Wp;      // packed fragments: [src][4 ksteps][8 ntiles][32 lanes][16]
  const float* bias;       // [128]
  float* Y;
  const float* rowscale;   // nullptr or [M]
  const float* exw;        // extra weight rows (f32) or ttab
  const float* u0;
  const float* xg;
  const float* tg;
  int nt;                  // time extent for row decode (64 main, 1 lift)
  int map1, map2;          // 0=identity 1=shift-x 2=shift-t 3=bcast-over-t
  int j;                   // shift
  int extra;               // 0 none, 1 sp(relx,du,|du|), 2 tp(relt), 3 ttab
  int act;                 // 0 none, 1 gelu
  int accum;               // 0 overwrite, 1 add into Y
};

__device__ __forceinline__ long maprow(int mode, long gm, int j, int nt) {
  if (mode == 0) return gm;
  long i = gm & 255;
  long rest = gm >> 8;                 // = b*nt + tt
  if (mode == 1) {
    long ii = i + j; ii = ii < 0 ? 0 : (ii > 255 ? 255 : ii);
    return (rest << 8) + ii;
  }
  if (mode == 2) {
    long tt = rest % nt, b = rest / nt;
    long tj = tt + j; tj = tj < 0 ? 0 : (tj >= nt ? nt - 1 : tj);
    return ((b * nt + tj) << 8) + i;
  }
  long b = rest / nt;                  // mode 3: X is [B*NX,128]
  return (b << 8) + i;
}

__global__ __launch_bounds__(256) void k_gemm128(GemmArgs A) {
  const int lane = threadIdx.x & 31;
  const int wv = threadIdx.x >> 5;
  const long rowBase = (long)blockIdx.x * 128 + wv * 16;
  const long gm_load = rowBase + (lane & 15);
  const int cb4 = (lane < 16) ? 0 : 2;     // col base 0 or 8 floats, in float4 units

  v8f acc[8];
#pragma unroll
  for (int tn = 0; tn < 8; ++tn) { v8f z = {}; acc[tn] = z; }

  const int nsrc = (A.X2 != nullptr) ? 2 : 1;
  for (int sidx = 0; sidx < nsrc; ++sidx) {
    const float* Xs = (sidx == 0) ? A.X1 : A.X2;
    const int mm = (sidx == 0) ? A.map1 : A.map2;
    long r = maprow(mm, gm_load, A.j, A.nt);
    const float4* xr = (const float4*)(Xs + r * 128);
    v16h af[4];
#pragma unroll
    for (int s = 0; s < 4; ++s) {
      float4 p0 = xr[s * 8 + cb4 + 0];
      float4 p1 = xr[s * 8 + cb4 + 1];
      float4 p2 = xr[s * 8 + cb4 + 4];
      float4 p3 = xr[s * 8 + cb4 + 5];
      af[s][0]  = (_Float16)p0.x; af[s][1]  = (_Float16)p0.y;
      af[s][2]  = (_Float16)p0.z; af[s][3]  = (_Float16)p0.w;
      af[s][4]  = (_Float16)p1.x; af[s][5]  = (_Float16)p1.y;
      af[s][6]  = (_Float16)p1.z; af[s][7]  = (_Float16)p1.w;
      af[s][8]  = (_Float16)p2.x; af[s][9]  = (_Float16)p2.y;
      af[s][10] = (_Float16)p2.z; af[s][11] = (_Float16)p2.w;
      af[s][12] = (_Float16)p3.x; af[s][13] = (_Float16)p3.y;
      af[s][14] = (_Float16)p3.z; af[s][15] = (_Float16)p3.w;
    }
    const _Float16* wp = A.Wp + (size_t)sidx * 16384;
#pragma unroll
    for (int s = 0; s < 4; ++s) {
#pragma unroll
      for (int tn = 0; tn < 8; ++tn) {
        v16h bf = *(const v16h*)(wp + (size_t)(s * 8 + tn) * 512 + (size_t)lane * 16);
        acc[tn] = __builtin_amdgcn_wmma_f32_16x16x32_f16(
            false, af[s], false, bf, (short)0, acc[tn], false, false);
      }
    }
  }

  // ------------------------------ epilogue ---------------------------------
  const int n0 = lane & 15;
  const long eBase = rowBase + ((long)(lane >> 4) << 3);
  float bn[8];
#pragma unroll
  for (int tn = 0; tn < 8; ++tn) bn[tn] = A.bias ? A.bias[tn * 16 + n0] : 0.f;

  float ew0[8], ew1[8], ew2[8];
  if (A.extra == 1) {
#pragma unroll
    for (int tn = 0; tn < 8; ++tn) {
      int n = tn * 16 + n0;
      ew0[tn] = A.exw[n]; ew1[tn] = A.exw[128 + n]; ew2[tn] = A.exw[256 + n];
    }
  } else if (A.extra == 2) {
#pragma unroll
    for (int tn = 0; tn < 8; ++tn) ew0[tn] = A.exw[tn * 16 + n0];
  }

  for (int e = 0; e < 8; ++e) {
    long gm = eBase + e;
    int i = (int)(gm & 255);
    long rest = gm >> 8;
    int tt = (int)(rest % A.nt);
    int b = (int)(rest / A.nt);
    float f0 = 0.f, f1 = 0.f, f2 = 0.f;
    if (A.extra == 1) {
      int ij = i + A.j; ij = ij < 0 ? 0 : (ij > 255 ? 255 : ij);
      float ui = A.u0[b * 256 + i], uj = A.u0[b * 256 + ij];
      f0 = A.xg[ij] - A.xg[i];
      f1 = ui - uj;
      f2 = fabsf(f1);
    } else if (A.extra == 2) {
      int tj = tt + A.j; tj = tj < 0 ? 0 : (tj >= A.nt ? A.nt - 1 : tj);
      f0 = A.tg[tj] - A.tg[tt];
    }
    float rs = A.rowscale ? A.rowscale[gm] : 1.f;
    float* yrow = A.Y + gm * 128;
#pragma unroll
    for (int tn = 0; tn < 8; ++tn) {
      int n = tn * 16 + n0;
      float v = acc[tn][e] + bn[tn];
      if (A.extra == 1)      v += f0 * ew0[tn] + f1 * ew1[tn] + f2 * ew2[tn];
      else if (A.extra == 2) v += f0 * ew0[tn];
      else if (A.extra == 3) v += A.exw[tt * 128 + n];
      if (A.act == 1) v = gelu_f(v);
      v *= rs;
      if (A.accum) v += yrow[n];
      yrow[n] = v;
    }
  }
}

// ------------------------------- driver ------------------------------------
extern "C" void kernel_launch(void* const* d_in, const int* in_sizes, int n_in,
                              void* d_out, int out_size, void* d_ws, size_t ws_size,
                              hipStream_t stream) {
  (void)in_sizes; (void)n_in; (void)out_size; (void)ws_size;
  auto F = [&](int idx) -> const float* { return (const float*)d_in[idx]; };

  const float* u0 = F(0);
  const float* xg = F(1);
  const float* tg = F(2);

  const long Mbig = (long)BD * NTD * NXD;   // 65536
  const long Mlft = (long)BD * NXD;         // 1024

  // ---- workspace bump allocator ----
  char* base = (char*)d_ws;
  size_t off = 0;
  auto af32 = [&](size_t n) -> float* {
    float* p = (float*)(base + off);
    off += ((n * 4 + 255) & ~(size_t)255);
    return p;
  };
  auto af16 = [&](size_t n) -> _Float16* {
    _Float16* p = (_Float16*)(base + off);
    off += ((n * 2 + 255) & ~(size_t)255);
    return p;
  };

  float* h    = af32((size_t)Mbig * 128);
  float* agg  = af32((size_t)Mbig * 128);
  float* z1   = af32((size_t)Mbig * 128);
  float* z2   = af32((size_t)Mbig * 128);
  float* g    = af32((size_t)Mbig);
  float* hsp  = af32((size_t)Mlft * 128);
  float* tA   = af32((size_t)Mlft * 128);
  float* tB   = af32((size_t)Mlft * 128);
  float* hself= af32((size_t)Mlft * 128);
  float* aggL = af32((size_t)Mlft * 128);
  float* gL   = af32((size_t)Mlft);
  float* tau  = af32(NTD * 32);
  float* ttab = af32(NTD * 128);

  const size_t PK1 = 16384, PK2 = 32768;
  _Float16* wp_ln2  = af16(PK1);
  _Float16* wp_le2  = af16(PK1);
  _Float16* wp_lc1  = af16(PK2);
  _Float16* wp_lc2  = af16(PK1);
  _Float16* wp_tm1  = af16(PK1);
  _Float16* wp_tm2  = af16(PK1);
  _Float16* wp_dec1 = af16(PK1);
  _Float16* wp_dec2 = af16(PK1);
  _Float16 *wp_spm1[3], *wp_spm2[3], *wp_spm3[3], *wp_spg1[3];
  _Float16 *wp_tpm1[3], *wp_tpm2[3], *wp_tpm3[3], *wp_tpg1[3];
  _Float16 *wp_upd1[3], *wp_upd2[3], *wp_upd3[3], *wp_WW[3];
  for (int l = 0; l < 3; ++l) {
    wp_spm1[l] = af16(PK2); wp_spm2[l] = af16(PK1); wp_spm3[l] = af16(PK1);
    wp_spg1[l] = af16(PK2);
    wp_tpm1[l] = af16(PK2); wp_tpm2[l] = af16(PK1); wp_tpm3[l] = af16(PK1);
    wp_tpg1[l] = af16(PK2);
    wp_upd1[l] = af16(PK2); wp_upd2[l] = af16(PK1); wp_upd3[l] = af16(PK1);
    wp_WW[l]   = af16(PK1);
  }

  auto PACK = [&](int widx, _Float16* dst, int Ktot) {
    int n = Ktot * 128;
    k_pack<<<(n + 255) / 256, 256, 0, stream>>>(F(widx), dst, Ktot);
  };
  auto GEMM = [&](const float* X1, int map1, const float* X2, int map2,
                  const _Float16* Wp, const float* bias, float* Y, long M, int nt,
                  int j, int extra, const float* exw, int act, int accum,
                  const float* rowscale) {
    GemmArgs a;
    a.X1 = X1; a.X2 = X2; a.Wp = Wp; a.bias = bias; a.Y = Y;
    a.rowscale = rowscale; a.exw = exw; a.u0 = u0; a.xg = xg; a.tg = tg;
    a.nt = nt; a.map1 = map1; a.map2 = map2; a.j = j;
    a.extra = extra; a.act = act; a.accum = accum;
    k_gemm128<<<(int)(M / 128), 256, 0, stream>>>(a);
  };

  // ---- pack all 128-wide weight matrices into fragment order ----
  PACK(5,  wp_ln2, 128);   // lift_node L2
  PACK(9,  wp_le2, 128);   // lift_edge L2
  PACK(15, wp_lc1, 256);   // lift_comb L1 (256-K)
  PACK(17, wp_lc2, 128);   // lift_comb L2
  PACK(22, wp_tm1, 128);   // time_mix L1, rows 0..127 (hsp part)
  PACK(24, wp_tm2, 128);   // time_mix L2
  PACK(26, wp_dec1, 128);
  PACK(28, wp_dec2, 128);
  for (int l = 0; l < 3; ++l) {
    int bI = 32 + 28 * l;
    PACK(bI + 0,  wp_spm1[l], 256);
    PACK(bI + 2,  wp_spm2[l], 128);
    PACK(bI + 4,  wp_spm3[l], 128);
    PACK(bI + 6,  wp_spg1[l], 256);
    PACK(bI + 10, wp_tpm1[l], 256);
    PACK(bI + 12, wp_tpm2[l], 128);
    PACK(bI + 14, wp_tpm3[l], 128);
    PACK(bI + 16, wp_tpg1[l], 256);
    PACK(bI + 20, wp_upd1[l], 256);
    PACK(bI + 22, wp_upd2[l], 128);
    PACK(bI + 24, wp_upd3[l], 128);
    PACK(bI + 26, wp_WW[l],   128);
  }

  // ------------------------------ lift stage -------------------------------
  k_lift_node1<<<(int)(Mlft * 128 / 256), 256, 0, stream>>>(u0, xg, F(3), F(4), tA);
  GEMM(tA, 0, nullptr, 0, wp_ln2, F(6), hself, Mlft, 1, 0, 0, nullptr, 0, 0, nullptr);
  k_zero<<<(int)((Mlft * 128 + 255) / 256), 256, 0, stream>>>(aggL, Mlft * 128);
  for (int j = -3; j <= 3; ++j) {
    k_lift_ein1<<<(int)(Mlft * 128 / 256), 256, 0, stream>>>(u0, xg, F(7),  F(8),  tA, j);
    k_lift_ein1<<<(int)(Mlft * 128 / 256), 256, 0, stream>>>(u0, xg, F(11), F(12), tB, j);
    k_rowdot<<<(int)(Mlft / 8), 256, 0, stream>>>(tB, F(13), F(14), nullptr, gL, Mlft, 0, 1);
    GEMM(tA, 0, nullptr, 0, wp_le2, F(10), aggL, Mlft, 1, 0, 0, nullptr, 0, 1, gL);
  }
  GEMM(hself, 0, aggL, 0, wp_lc1, F(16), tA,  Mlft, 1, 0, 0, nullptr, 1, 0, nullptr);
  GEMM(tA,    0, nullptr, 0, wp_lc2, F(18), hsp, Mlft, 1, 0, 0, nullptr, 0, 0, nullptr);

  // ------------------------------ time stage -------------------------------
  k_time_tau<<<8, 256, 0, stream>>>(tg, F(19), F(20), F(21), tau);
  k_ttab<<<32, 256, 0, stream>>>(tau, F(22), ttab);
  GEMM(hsp, 3, nullptr, 0, wp_tm1, F(23), z1, Mbig, NTD, 0, 3, ttab, 1, 0, nullptr);
  GEMM(z1,  0, nullptr, 0, wp_tm2, F(25), h,  Mbig, NTD, 0, 0, nullptr, 0, 0, nullptr);

  // ------------------------------ main layers ------------------------------
  for (int l = 0; l < 3; ++l) {
    int bI = 32 + 28 * l;
    const float* spm_ex = F(bI + 0) + 256 * 128;
    const float* spg_ex = F(bI + 6) + 256 * 128;
    const float* tpm_ex = F(bI + 10) + 256 * 128;
    const float* tpg_ex = F(bI + 16) + 256 * 128;

    k_zero<<<(int)((Mbig * 128 + 255) / 256), 256, 0, stream>>>(agg, Mbig * 128);

    for (int j = -3; j <= 3; ++j) {
      GEMM(h, 0, h, 1, wp_spg1[l], F(bI + 7), z1, Mbig, NTD, j, 1, spg_ex, 1, 0, nullptr);
      k_rowdot<<<(int)(Mbig / 8), 256, 0, stream>>>(z1, F(bI + 8), F(bI + 9), nullptr, g, Mbig, 0, NTD);
      GEMM(h,  0, h, 1, wp_spm1[l], F(bI + 1), z1,  Mbig, NTD, j, 1, spm_ex, 1, 0, nullptr);
      GEMM(z1, 0, nullptr, 0, wp_spm2[l], F(bI + 3), z2,  Mbig, NTD, 0, 0, nullptr, 1, 0, nullptr);
      GEMM(z2, 0, nullptr, 0, wp_spm3[l], F(bI + 5), agg, Mbig, NTD, 0, 0, nullptr, 0, 1, g);
    }
    for (int j = -2; j <= 2; ++j) {
      GEMM(h, 0, h, 2, wp_tpg1[l], F(bI + 17), z1, Mbig, NTD, j, 2, tpg_ex, 1, 0, nullptr);
      k_rowdot<<<(int)(Mbig / 8), 256, 0, stream>>>(z1, F(bI + 18), F(bI + 19), nullptr, g, Mbig, 0, NTD);
      GEMM(h,  0, h, 2, wp_tpm1[l], F(bI + 11), z1,  Mbig, NTD, j, 2, tpm_ex, 1, 0, nullptr);
      GEMM(z1, 0, nullptr, 0, wp_tpm2[l], F(bI + 13), z2,  Mbig, NTD, 0, 0, nullptr, 1, 0, nullptr);
      GEMM(z2, 0, nullptr, 0, wp_tpm3[l], F(bI + 15), agg, Mbig, NTD, 0, 0, nullptr, 0, 1, g);
    }
    // upd MLP on [h, agg], then residual W branch, then gelu
    GEMM(h,  0, agg, 0, wp_upd1[l], F(bI + 21), z1, Mbig, NTD, 0, 0, nullptr, 1, 0, nullptr);
    GEMM(z1, 0, nullptr, 0, wp_upd2[l], F(bI + 23), z2, Mbig, NTD, 0, 0, nullptr, 1, 0, nullptr);
    GEMM(z2, 0, nullptr, 0, wp_upd3[l], F(bI + 25), z1, Mbig, NTD, 0, 0, nullptr, 0, 0, nullptr);
    GEMM(h,  0, nullptr, 0, wp_WW[l],   F(bI + 27), z1, Mbig, NTD, 0, 0, nullptr, 0, 1, nullptr);
    k_gelu_ew<<<(int)((Mbig * 128 + 255) / 256), 256, 0, stream>>>(z1, h, Mbig * 128);
  }

  // ------------------------------- decoder ---------------------------------
  GEMM(h,  0, nullptr, 0, wp_dec1, F(27), z1, Mbig, NTD, 0, 0, nullptr, 1, 0, nullptr);
  GEMM(z1, 0, nullptr, 0, wp_dec2, F(29), z2, Mbig, NTD, 0, 0, nullptr, 1, 0, nullptr);
  k_rowdot<<<(int)(Mbig / 8), 256, 0, stream>>>(z2, F(30), F(31), u0, (float*)d_out,
                                                Mbig, 1, NTD);
}
